// Attention_80341658239275
// MI455X (gfx1250) — matrix operand order repaired and checked by reading, not compiled, not run on hardware
//
#include <hip/hip_runtime.h>
#include <hip/hip_bf16.h>

// ---------------------------------------------------------------------------
// Problem constants: B=4, N=1024, C=1024, H=16, HD=64, N_CAUSAL=8,
// SCALE = HD^-0.5 * 100 = 12.5
// ---------------------------------------------------------------------------
#define BB    4
#define NN    1024
#define CC    1024
#define HH    16
#define HD    64
#define NCAUS 8
#define SCALE 12.5f
#define LOG2E 1.44269504088896340736f

typedef __attribute__((ext_vector_type(16))) __bf16 v16bf;
typedef __attribute__((ext_vector_type(8)))  float  v8f;

union FragBF { uint4 q[2]; v16bf v; };

static __device__ __forceinline__ v8f v8f_zero() {
  v8f z;
#pragma unroll
  for (int i = 0; i < 8; ++i) z[i] = 0.0f;
  return z;
}

static __device__ __forceinline__ unsigned short f2bf_bits(float f) {
  __bf16 v = (__bf16)f;
  unsigned short u;
  __builtin_memcpy(&u, &v, 2);
  return u;
}

// --- CDNA5 async global->LDS copy (ASYNCcnt-tracked, no VGPR data path) ----
static __device__ __forceinline__ void async_b128(void* lds, const void* g) {
  unsigned lo = (unsigned)(unsigned long long)(uintptr_t)lds;  // LDS offset
  asm volatile("global_load_async_to_lds_b128 %0, %1, off"
               :: "v"(lo), "v"((unsigned long long)(uintptr_t)g)
               : "memory");
}
#define ASYNC_WAIT(imm) asm volatile("s_wait_asynccnt " #imm ::: "memory")

// ---------------------------------------------------------------------------
// Kernel 1a: f32 -> bf16 conversion (grid-stride)
// ---------------------------------------------------------------------------
__global__ void cvt_f32_bf16(const float* __restrict__ src,
                             __bf16* __restrict__ dst, int n) {
  int i = blockIdx.x * blockDim.x + threadIdx.x;
  int stride = gridDim.x * blockDim.x;
  for (; i < n; i += stride) dst[i] = (__bf16)src[i];
}

// ---------------------------------------------------------------------------
// Kernel 1b: f32 [Nn,K] -> bf16 transposed [K,Nn]  (LDS-tiled, coalesced)
// Weights are transposed ONCE so both GEMMs become NN-GEMMs with contiguous
// B tiles (async-copyable, no per-iteration scalar LDS transpose).
// ---------------------------------------------------------------------------
__global__ void cvt_transpose_bf16(const float* __restrict__ src,  // [Nn,K]
                                   __bf16* __restrict__ dst,       // [K,Nn]
                                   int Nn, int K) {
  __shared__ __bf16 tile[32][33];
  const int kBase = blockIdx.x * 32;
  const int nBase = blockIdx.y * 32;
  const int tx = threadIdx.x & 31;
  const int ty = threadIdx.x >> 5;  // 0..7
#pragma unroll
  for (int i = 0; i < 32; i += 8)
    tile[ty + i][tx] = (__bf16)src[(size_t)(nBase + ty + i) * K + kBase + tx];
  __syncthreads();
#pragma unroll
  for (int i = 0; i < 32; i += 8)
    dst[(size_t)(kBase + ty + i) * Nn + nBase + tx] = tile[tx][ty + i];
}

// ---------------------------------------------------------------------------
// Kernel 2/4: NN GEMM  Y[m,n] = sum_k A[m,k] * Bt[k,n] + bias[n]
//   A: bf16 [M,K] row-major,  Bt: bf16 [K,Nn] row-major (pre-transposed W)
//   mode 0: scatter q -> [B,H,N,HD], k -> [B,H,HD,N] (transposed!), v -> [B,H,N,HD]
//   mode 1: write f32 row-major [M,Nn]
// Tiles: 128x128x32; 256 threads = 8 waves; wave computes 32x64.
// Double-buffered LDS, tiles staged with global_load_async_to_lds_b128.
// ---------------------------------------------------------------------------
#define BM 128
#define BN 128
#define BK 32
#define LDA_P 40   // halves pitch, A tile rows   (80B)
#define LDB_P 136  // halves pitch, Bt tile rows  (272B)

__global__ __launch_bounds__(256)
void gemm_nn_bf16(const __bf16* __restrict__ A, const __bf16* __restrict__ Bt,
                  const float* __restrict__ bias, int M, int Nn, int K,
                  int mode,
                  __bf16* __restrict__ qb, __bf16* __restrict__ kbT,
                  __bf16* __restrict__ vb, float* __restrict__ outf) {
  __shared__ __align__(16) unsigned short sA[2][BM * LDA_P];
  __shared__ __align__(16) unsigned short sB[2][BK * LDB_P];

  const int tid  = threadIdx.x;
  const int lane = tid & 31;
  const int wave = tid >> 5;
  const int wm   = wave & 3;
  const int wn   = wave >> 2;
  const int g    = lane >> 4;
  const int l15  = lane & 15;

  const int mBase = blockIdx.y * BM;
  const int nBase = blockIdx.x * BN;

  v8f c[2][4];
#pragma unroll
  for (int fm = 0; fm < 2; ++fm)
#pragma unroll
    for (int fn = 0; fn < 4; ++fn) c[fm][fn] = v8f_zero();

  // staging roles
  const int aRow = tid >> 1, aCol = (tid & 1) * 16;  // A: 2 x b128 per thread
  const int bRow = tid >> 3, bCol = (tid & 7) * 16;  // Bt: 2 x b128 per thread

  const int ksteps = K / BK;

  // issue tile 0 into buffer 0
  {
    const __bf16* ga = A + (size_t)(mBase + aRow) * K + aCol;
    async_b128(&sA[0][aRow * LDA_P + aCol],     ga);
    async_b128(&sA[0][aRow * LDA_P + aCol + 8], ga + 8);
    const __bf16* gb = Bt + (size_t)bRow * Nn + nBase + bCol;
    async_b128(&sB[0][bRow * LDB_P + bCol],     gb);
    async_b128(&sB[0][bRow * LDB_P + bCol + 8], gb + 8);
  }

  for (int i = 0; i < ksteps; ++i) {
    const int buf = i & 1;
    __syncthreads();  // readers of tile i-1 done -> buf^1 free
    if (i + 1 < ksteps) {
      const int k1 = (i + 1) * BK;
      const __bf16* ga = A + (size_t)(mBase + aRow) * K + k1 + aCol;
      async_b128(&sA[buf ^ 1][aRow * LDA_P + aCol],     ga);
      async_b128(&sA[buf ^ 1][aRow * LDA_P + aCol + 8], ga + 8);
      const __bf16* gb = Bt + (size_t)(k1 + bRow) * Nn + nBase + bCol;
      async_b128(&sB[buf ^ 1][bRow * LDB_P + bCol],     gb);
      async_b128(&sB[buf ^ 1][bRow * LDB_P + bCol + 8], gb + 8);
      ASYNC_WAIT(0x4);  // in-order completion: tile i's 4 transfers landed
    } else {
      ASYNC_WAIT(0x0);
    }
    __syncthreads();  // everyone's tile-i transfers visible

    FragBF af[2];
#pragma unroll
    for (int fm = 0; fm < 2; ++fm) {
      int row = wm * 32 + fm * 16 + l15;
      int hi  = g * 8;
      af[fm].q[0] = *(const uint4*)&sA[buf][row * LDA_P + hi];
      af[fm].q[1] = *(const uint4*)&sA[buf][row * LDA_P + 16 + hi];
    }
#pragma unroll
    for (int fn = 0; fn < 4; ++fn) {
      FragBF bf;
      int col = wn * 64 + fn * 16;
#pragma unroll
      for (int s = 0; s < 16; ++s) {
        int kk = (s < 8) ? g * 8 + s : 16 + g * 8 + (s - 8);
        ((unsigned short*)&bf)[s] = sB[buf][kk * LDB_P + col + l15];
      }
#pragma unroll
      for (int fm = 0; fm < 2; ++fm)
        c[fm][fn] = __builtin_amdgcn_wmma_f32_16x16x32_bf16(
            false, af[fm].v, false, bf.v, (short)0, c[fm][fn], false, false);
    }
  }

  // ---- epilogue ----
#pragma unroll
  for (int fm = 0; fm < 2; ++fm) {
#pragma unroll
    for (int fn = 0; fn < 4; ++fn) {
      int n = nBase + wn * 64 + fn * 16 + l15;
      float bv = bias[n];
#pragma unroll
      for (int r = 0; r < 8; ++r) {
        int m = mBase + wm * 32 + fm * 16 + r + g * 8;
        float val = c[fm][fn][r] + bv;
        if (mode == 0) {
          int which = n >> 10;
          int cc2 = n & 1023;
          int h = cc2 >> 6, d = cc2 & 63;
          int b = m >> 10, row = m & 1023;
          if (which == 0) {
            qb[(((size_t)b * HH + h) * NN + row) * HD + d] = (__bf16)val;
          } else if (which == 1) {
            // store K transposed: [B,H,HD,N] so attention stages it contiguously
            kbT[(((size_t)b * HH + h) * HD + d) * NN + row] = (__bf16)val;
          } else {
            vb[(((size_t)b * HH + h) * NN + row) * HD + d] = (__bf16)val;
          }
        } else {
          outf[(size_t)m * Nn + n] = val;
        }
      }
    }
  }
}

// ---------------------------------------------------------------------------
// Kernel 3: flash attention. Grid (N/128, B*H); 8 waves x 16 rows each.
// K^T chunks [64 d][32 m] staged via async copy, double-buffered.
// ---------------------------------------------------------------------------
#define KT_P 40
#define P_P  40

__global__ __launch_bounds__(256)
void attn_flash(const __bf16* __restrict__ Q,    // [B,H,N,HD]
                const __bf16* __restrict__ KT,   // [B,H,HD,N]  (transposed)
                const __bf16* __restrict__ V,    // [B,H,N,HD]
                const float* __restrict__ diag,  // [B, 8]
                const float* __restrict__ band,  // [8, N, N]
                const float* __restrict__ noise, // [B, H, N, N]
                const unsigned char* __restrict__ mask,  // bool [B,H,N,N]
                __bf16* __restrict__ O)          // [B, N, C] (h*64+d)
{
  __shared__ __align__(16) unsigned short sKt[2][64 * KT_P];  // [d][m]
  __shared__ __align__(16) unsigned short sP[8 * 16 * P_P];

  const int tid  = threadIdx.x;
  const int lane = tid & 31;
  const int wave = tid >> 5;
  const int g    = lane >> 4;
  const int l15  = lane & 15;

  const int bh = blockIdx.y;
  const int b  = bh >> 4, h = bh & 15;
  const int r0 = blockIdx.x * 128 + wave * 16;
  const size_t base = (size_t)bh * NN * HD;  // 64K elems per (b,h)

  // Q fragments held across the whole loop
  FragBF qa[2];
  {
    const __bf16* qp = Q + base + (size_t)(r0 + l15) * HD;
    int hi = g * 8;
    qa[0].q[0] = *(const uint4*)(qp + hi);
    qa[0].q[1] = *(const uint4*)(qp + 16 + hi);
    qa[1].q[0] = *(const uint4*)(qp + 32 + hi);
    qa[1].q[1] = *(const uint4*)(qp + 48 + hi);
  }

  float m_run[8], l_run[8];
  v8f o[4];
#pragma unroll
  for (int r = 0; r < 8; ++r) { m_run[r] = -1e30f; l_run[r] = 0.0f; }
#pragma unroll
  for (int dg = 0; dg < 4; ++dg) o[dg] = v8f_zero();

  // staging role: KT rows d = tid>>2 (0..63), 16B part = tid&3
  const int sd = tid >> 2, sp = (tid & 3) * 8;

  // prologue: stage chunk 0
  async_b128(&sKt[0][sd * KT_P + sp], KT + base + (size_t)sd * NN + sp);

  for (int j = 0; j < 32; ++j) {
    const int m0 = j * 32;
    const int buf = j & 1;
    __syncthreads();  // chunk j-1 consumers done
    if (j + 1 < 32) {
      async_b128(&sKt[buf ^ 1][sd * KT_P + sp],
                 KT + base + (size_t)sd * NN + (m0 + 32) + sp);
      ASYNC_WAIT(0x1);  // chunk j landed (in-order)
      __builtin_prefetch(V + base + (size_t)(m0 + 32 + lane) * HD, 0, 0);
    } else {
      ASYNC_WAIT(0x0);
    }
    __syncthreads();

    bool active = (h >= NCAUS) || (m0 <= r0 + 15);
    if (active) {
      // ---- scores S = Q @ K^T over this 32-col chunk ----
      FragBF kf[2][2];
#pragma unroll
      for (int kc = 0; kc < 2; ++kc)
#pragma unroll
        for (int fn = 0; fn < 2; ++fn) {
#pragma unroll
          for (int s = 0; s < 16; ++s) {
            int kk = (s < 8) ? g * 8 + s : 16 + g * 8 + (s - 8);
            ((unsigned short*)&kf[kc][fn])[s] = sKt[buf][(kc * 32 + kk) * KT_P + fn * 16 + l15];
          }
        }
      v8f s[2];
#pragma unroll
      for (int fn = 0; fn < 2; ++fn) {
        s[fn] = v8f_zero();
        s[fn] = __builtin_amdgcn_wmma_f32_16x16x32_bf16(
            false, qa[0].v, false, kf[0][fn].v, (short)0, s[fn], false, false);
        s[fn] = __builtin_amdgcn_wmma_f32_16x16x32_bf16(
            false, qa[1].v, false, kf[1][fn].v, (short)0, s[fn], false, false);
      }

      // ---- scale + bias + masked noise, row max ----
      float rmax[8];
#pragma unroll
      for (int r = 0; r < 8; ++r) {
        int row = r0 + r + g * 8;
#pragma unroll
        for (int fn = 0; fn < 2; ++fn) {
          int col = m0 + fn * 16 + l15;
          float sv = s[fn][r] * SCALE;
          if (h < NCAUS) {
            if (col > row)       sv = -1e30f;
            else if (col == row) sv += diag[b * NCAUS + h];
          } else {
            sv += band[((size_t)(h - NCAUS) * NN + row) * NN + col];
          }
          size_t idx = ((size_t)bh * NN + row) * NN + col;
          if (mask[idx]) sv += noise[idx];
          s[fn][r] = sv;
        }
        rmax[r] = fmaxf(s[0][r], s[1][r]);
      }
#pragma unroll
      for (int r = 0; r < 8; ++r) {
        rmax[r] = fmaxf(rmax[r], __shfl_xor(rmax[r], 1, 32));
        rmax[r] = fmaxf(rmax[r], __shfl_xor(rmax[r], 2, 32));
        rmax[r] = fmaxf(rmax[r], __shfl_xor(rmax[r], 4, 32));
        rmax[r] = fmaxf(rmax[r], __shfl_xor(rmax[r], 8, 32));
      }

      // ---- online softmax ----
      float alpha[8], rsum[8];
#pragma unroll
      for (int r = 0; r < 8; ++r) {
        float mn = fmaxf(m_run[r], rmax[r]);
        alpha[r] = exp2f((m_run[r] - mn) * LOG2E);
        m_run[r] = mn;
        float acc = 0.0f;
#pragma unroll
        for (int fn = 0; fn < 2; ++fn) {
          float p = exp2f((s[fn][r] - m_run[r]) * LOG2E);
          s[fn][r] = p;
          acc += p;
        }
        rsum[r] = acc;
      }
#pragma unroll
      for (int r = 0; r < 8; ++r) {
        rsum[r] += __shfl_xor(rsum[r], 1, 32);
        rsum[r] += __shfl_xor(rsum[r], 2, 32);
        rsum[r] += __shfl_xor(rsum[r], 4, 32);
        rsum[r] += __shfl_xor(rsum[r], 8, 32);
        l_run[r] = l_run[r] * alpha[r] + rsum[r];
      }
#pragma unroll
      for (int dg = 0; dg < 4; ++dg)
#pragma unroll
        for (int r = 0; r < 8; ++r) o[dg][r] *= alpha[r];

      // ---- P: C-layout -> A-layout via per-wave LDS round trip ----
      unsigned short* myP = &sP[wave * 16 * P_P];
#pragma unroll
      for (int r = 0; r < 8; ++r) {
        int prow = r + g * 8;
        myP[prow * P_P + l15]      = f2bf_bits(s[0][r]);
        myP[prow * P_P + 16 + l15] = f2bf_bits(s[1][r]);
      }
      __threadfence_block();
      FragBF pa;
      pa.q[0] = *(const uint4*)&myP[l15 * P_P + g * 8];
      pa.q[1] = *(const uint4*)&myP[l15 * P_P + 16 + g * 8];

      // ---- O += P @ V ----
      const __bf16* vrow = V + base + (size_t)m0 * HD;
#pragma unroll
      for (int dg = 0; dg < 4; ++dg) {
        FragBF vf;
#pragma unroll
        for (int s = 0; s < 16; ++s) {
          int kk = (s < 8) ? g * 8 + s : 16 + g * 8 + (s - 8);
          vf.v[s] = vrow[(size_t)kk * HD + dg * 16 + l15];
        }
        o[dg] = __builtin_amdgcn_wmma_f32_16x16x32_bf16(
            false, pa.v, false, vf.v, (short)0, o[dg], false, false);
      }
    }
  }

  // ---- normalize, store bf16 [B, N, h*64+d] ----
  float inv[8];
#pragma unroll
  for (int r = 0; r < 8; ++r) inv[r] = 1.0f / l_run[r];
#pragma unroll
  for (int dg = 0; dg < 4; ++dg) {
#pragma unroll
    for (int r = 0; r < 8; ++r) {
      int row = r0 + r + g * 8;
      int d = dg * 16 + l15;
      O[((size_t)b * NN + row) * CC + h * HD + d] = (__bf16)(o[dg][r] * inv[r]);
    }
  }
}

// ---------------------------------------------------------------------------
// Host launcher
// ---------------------------------------------------------------------------
extern "C" void kernel_launch(void* const* d_in, const int* in_sizes, int n_in,
                              void* d_out, int out_size, void* d_ws,
                              size_t ws_size, hipStream_t stream) {
  const float* x       = (const float*)d_in[0];
  const float* qkv_w   = (const float*)d_in[1];
  const float* qkv_b   = (const float*)d_in[2];
  const float* proj_w  = (const float*)d_in[3];
  const float* proj_b  = (const float*)d_in[4];
  const float* diag    = (const float*)d_in[5];
  const float* band    = (const float*)d_in[6];
  const float* noise   = (const float*)d_in[7];
  const unsigned char* mask = (const unsigned char*)d_in[8];
  float* out = (float*)d_out;

  char* ws = (char*)d_ws;
  size_t off = 0;
  __bf16* xb     = (__bf16*)(ws + off); off += (size_t)BB * NN * CC * 2;      // 8MB
  __bf16* wqkvT  = (__bf16*)(ws + off); off += (size_t)3 * CC * CC * 2;       // 6MB  [K,3C]
  __bf16* wprojT = (__bf16*)(ws + off); off += (size_t)CC * CC * 2;           // 2MB  [K,C]
  __bf16* qb     = (__bf16*)(ws + off); off += (size_t)BB * HH * NN * HD * 2; // 8MB
  __bf16* kbT    = (__bf16*)(ws + off); off += (size_t)BB * HH * NN * HD * 2; // 8MB [B,H,HD,N]
  __bf16* vb     = (__bf16*)(ws + off); off += (size_t)BB * HH * NN * HD * 2; // 8MB
  __bf16* ob     = (__bf16*)(ws + off); off += (size_t)BB * NN * CC * 2;      // 8MB

  // 1) convert x; convert+transpose weights (one-time, amortized)
  cvt_f32_bf16<<<4096, 256, 0, stream>>>(x, xb, BB * NN * CC);
  cvt_transpose_bf16<<<dim3(CC / 32, 3 * CC / 32), 256, 0, stream>>>(
      qkv_w, wqkvT, 3 * CC, CC);
  cvt_transpose_bf16<<<dim3(CC / 32, CC / 32), 256, 0, stream>>>(
      proj_w, wprojT, CC, CC);

  // 2) QKV projection -> q [B,H,N,HD], k^T [B,H,HD,N], v [B,H,N,HD]
  gemm_nn_bf16<<<dim3(3 * CC / BN, BB * NN / BM), 256, 0, stream>>>(
      xb, wqkvT, qkv_b, BB * NN, 3 * CC, CC, /*mode=*/0, qb, kbT, vb, nullptr);

  // 3) flash attention -> ob [B,N,C] bf16
  attn_flash<<<dim3(NN / 128, BB * HH), 256, 0, stream>>>(
      qb, kbT, vb, diag, band, noise, mask, ob);

  // 4) output projection -> f32 d_out
  gemm_nn_bf16<<<dim3(CC / BN, BB * NN / BM), 256, 0, stream>>>(
      ob, wprojT, proj_b, BB * NN, CC, CC, /*mode=*/1, nullptr, nullptr,
      nullptr, out);
}
